// ScaledConvolutionalDotProduct_57062935495338
// MI455X (gfx1250) — matrix operand order repaired and checked by reading, not compiled, not run on hardware
//
#include <hip/hip_runtime.h>
#include <hip/hip_bf16.h>

// Problem constants (reference: B=4, S=2048, D_IN=D_MODEL=1024)
#define BATCH 4
#define S_LEN 2048
#define D_DIM 1024
#define M_TOT (BATCH * S_LEN)   // 8192 rows for the projection GEMMs
#define TILE_ELEMS (128 * 32)   // one 128x32 f16 tile in LDS

typedef __attribute__((ext_vector_type(16))) _Float16 v16h;
typedef __attribute__((ext_vector_type(8)))  float    v8f;
typedef __attribute__((ext_vector_type(4)))  int      v4i;

union FragU { uint4 u[2]; v16h v; };

// ---------------------------------------------------------------------------
// CDNA5 async global->LDS copy (ASYNCcnt-tracked), 16 bytes per lane.
// Builtin signature (leaked by clang diagnostic): param0 = int4 addrspace(1)*.
// ---------------------------------------------------------------------------
static __device__ inline void async_copy16(const _Float16* g, _Float16* l) {
#if __has_builtin(__builtin_amdgcn_global_load_async_to_lds_b128)
  typedef __attribute__((address_space(1))) v4i gbuf_t;
  typedef __attribute__((address_space(3))) v4i lbuf_t;
  __builtin_amdgcn_global_load_async_to_lds_b128(
      (gbuf_t*)(void*)g, (lbuf_t*)(void*)l, 0, 0);
#else
  unsigned lds_off = (unsigned)(uintptr_t)l;   // low 32 bits of flat LDS addr
  asm volatile("global_load_async_to_lds_b128 %0, %1, off"
               :: "v"(lds_off), "v"(g) : "memory");
#endif
}

template <int N>
static __device__ inline void wait_asynccnt() {
#if __has_builtin(__builtin_amdgcn_s_wait_asynccnt)
  __builtin_amdgcn_s_wait_asynccnt((unsigned short)N);
#else
  asm volatile("s_wait_asynccnt %0" :: "i"(N) : "memory");
#endif
}

// Issue async copies for a 128x32 f16 tile (rows contiguous in K). 256 thr,
// 2 x b128 per thread => 2 async instructions per wave per tile.
static __device__ inline void async_tile(_Float16* dst_lds, const _Float16* src,
                                         int ldsrc, int tid) {
#pragma unroll
  for (int i = 0; i < 2; ++i) {
    int q = tid + i * 256;          // 512 x 16B total
    int r = q >> 2;
    int c = (q & 3) << 3;
    async_copy16(src + (size_t)r * ldsrc + c, dst_lds + r * 32 + c);
  }
}

// ---------------------------------------------------------------------------
// WMMA fragment loads from LDS (wave32 layouts per CDNA5 ISA 7.12.2).
// A tile: row-major [128][32] f16. Lane (m=lane%16, h=lane/16):
//   VGPR0-3 K=8h..8h+7, VGPR4-7 K=16+8h..16+8h+7.
// B tile: n-major [128][32] f16 (B^T). Lane (n=lane%16, h): same K as A.
// ---------------------------------------------------------------------------
static __device__ inline v16h load_frag_a(const _Float16* As, int row, int h) {
  FragU f;
  f.u[0] = *(const uint4*)(As + row * 32 + 8 * h);
  f.u[1] = *(const uint4*)(As + row * 32 + 16 + 8 * h);
  return f.v;
}
static __device__ inline v16h load_frag_b(const _Float16* Bs, int col, int h) {
  FragU f;
  f.u[0] = *(const uint4*)(Bs + col * 32 + 8 * h);
  f.u[1] = *(const uint4*)(Bs + col * 32 + 16 + 8 * h);
  return f.v;
}

// One K=32 step: 8 v_wmma_f32_16x16x32_f16 per wave (wave tile 64x32).
static __device__ inline void mma_step(const _Float16* As, const _Float16* Bs,
                                       int wave_m, int wave_n, int lane16, int h,
                                       v8f acc[4][2]) {
  v16h a[4];
#pragma unroll
  for (int mt = 0; mt < 4; ++mt)
    a[mt] = load_frag_a(As, wave_m * 64 + mt * 16 + lane16, h);
#pragma unroll
  for (int nt = 0; nt < 2; ++nt) {
    v16h b = load_frag_b(Bs, wave_n * 32 + nt * 16 + lane16, h);
#pragma unroll
    for (int mt = 0; mt < 4; ++mt)
      acc[mt][nt] = __builtin_amdgcn_wmma_f32_16x16x32_f16(
          false, a[mt], false, b, (short)0, acc[mt][nt], false, false);
  }
}

// Double-buffered async GEMM mainloop. gA/gB point at tile origin (k = 0).
template <int KT>
static __device__ inline void gemm_body(const _Float16* gA, int lda,
                                        const _Float16* gB, int ldb,
                                        _Float16* As, _Float16* Bs,
                                        int tid, int wave_m, int wave_n,
                                        int lane16, int h, v8f acc[4][2]) {
  async_tile(As, gA, lda, tid);
  async_tile(Bs, gB, ldb, tid);
  for (int kt = 0; kt < KT; ++kt) {
    const int cur = kt & 1;
    __syncthreads();                      // buffer cur^1 free for refill
    if (kt + 1 < KT) {
      async_tile(As + (cur ^ 1) * TILE_ELEMS, gA + (kt + 1) * 32, lda, tid);
      async_tile(Bs + (cur ^ 1) * TILE_ELEMS, gB + (kt + 1) * 32, ldb, tid);
      wait_asynccnt<4>();                 // newest 4 pending => tile kt landed
    } else {
      wait_asynccnt<0>();
    }
    __syncthreads();                      // all waves' data for tile kt in LDS
    mma_step(As + cur * TILE_ELEMS, Bs + cur * TILE_ELEMS,
             wave_m, wave_n, lane16, h, acc);
  }
}

// ---------------------------------------------------------------------------
// Kernel: f32 -> f16 convert (streaming, 8 elems/thread)
// ---------------------------------------------------------------------------
__global__ __launch_bounds__(256) void cvt_f16_kernel(const float* __restrict__ src,
                                                      _Float16* __restrict__ dst) {
  size_t i = ((size_t)blockIdx.x * 256 + threadIdx.x) * 8;
  float4 a = *(const float4*)(src + i);
  float4 b = *(const float4*)(src + i + 4);
  union { uint4 u; _Float16 h[8]; } o;
  o.h[0] = (_Float16)a.x; o.h[1] = (_Float16)a.y;
  o.h[2] = (_Float16)a.z; o.h[3] = (_Float16)a.w;
  o.h[4] = (_Float16)b.x; o.h[5] = (_Float16)b.y;
  o.h[6] = (_Float16)b.z; o.h[7] = (_Float16)b.w;
  *(uint4*)(dst + i) = o.u;
}

// ---------------------------------------------------------------------------
// Kernel: transpose + f32->f16 one 1024x1024 weight: Wt[n][k] = W[k][n]
// ---------------------------------------------------------------------------
__global__ __launch_bounds__(256) void transpose_w_kernel(const float* __restrict__ W,
                                                          _Float16* __restrict__ Wt) {
  __shared__ _Float16 tile[32][33];
  int bx = blockIdx.x * 32;   // n tile
  int by = blockIdx.y * 32;   // k tile
  int tx = threadIdx.x;       // 0..31
  int ty = threadIdx.y;       // 0..7
#pragma unroll
  for (int i = 0; i < 32; i += 8)
    tile[ty + i][tx] = (_Float16)W[(size_t)(by + ty + i) * D_DIM + bx + tx];
  __syncthreads();
#pragma unroll
  for (int i = 0; i < 32; i += 8)
    Wt[(size_t)(bx + ty + i) * D_DIM + by + tx] = tile[tx][ty + i];
}

// ---------------------------------------------------------------------------
// Kernel 1: P = relu(Xh @ W + bias). Xh f16 [8192][1024], Wt f16 [N][K].
// TRANSPOSED==0: out f16 [8192][1024].  TRANSPOSED==1: vpT f16 [B][D][S].
// ---------------------------------------------------------------------------
template <int TRANSPOSED>
__global__ __launch_bounds__(256) void proj_kernel(const _Float16* __restrict__ Xh,
                                                   const _Float16* __restrict__ Wt,
                                                   const float* __restrict__ bias,
                                                   _Float16* __restrict__ out) {
  __shared__ _Float16 As[2 * TILE_ELEMS];
  __shared__ _Float16 Bs[2 * TILE_ELEMS];

  const int tid = threadIdx.x;
  const int wid = tid >> 5, lane = tid & 31;
  const int lane16 = lane & 15, h = lane >> 4;
  const int wave_m = wid & 1, wave_n = wid >> 1;
  const int m0 = blockIdx.y * 128;
  const int n0 = blockIdx.x * 128;

  v8f acc[4][2] = {};
  gemm_body<D_DIM / 32>(Xh + (size_t)m0 * D_DIM, D_DIM,
                        Wt + (size_t)n0 * D_DIM, D_DIM,
                        As, Bs, tid, wave_m, wave_n, lane16, h, acc);

#pragma unroll
  for (int mt = 0; mt < 4; ++mt)
#pragma unroll
    for (int nt = 0; nt < 2; ++nt) {
      int ncol = n0 + wave_n * 32 + nt * 16 + lane16;
      float bv = bias[ncol];
#pragma unroll
      for (int j = 0; j < 8; ++j) {
        int mrow = m0 + wave_m * 64 + mt * 16 + j + 8 * h;
        float val = fmaxf(acc[mt][nt][j] + bv, 0.0f);
        if (TRANSPOSED == 0) {
          out[(size_t)mrow * D_DIM + ncol] = (_Float16)val;
        } else {
          int b = mrow >> 11;          // / S_LEN
          int s = mrow & (S_LEN - 1);
          out[((size_t)b * D_DIM + ncol) * S_LEN + s] = (_Float16)val;
        }
      }
    }
}

// ---------------------------------------------------------------------------
// Kernel 2: logits[b][sq][sk] = relu(qp[b] @ kp[b]^T)/32 + mask[sq][sk]
// ---------------------------------------------------------------------------
__global__ __launch_bounds__(256) void scores_kernel(const _Float16* __restrict__ qp,
                                                     const _Float16* __restrict__ kp,
                                                     const float* __restrict__ mask,
                                                     _Float16* __restrict__ logits) {
  __shared__ _Float16 As[2 * TILE_ELEMS];
  __shared__ _Float16 Bs[2 * TILE_ELEMS];

  const int tid = threadIdx.x;
  const int wid = tid >> 5, lane = tid & 31;
  const int lane16 = lane & 15, h = lane >> 4;
  const int wave_m = wid & 1, wave_n = wid >> 1;
  const int b = blockIdx.z;
  const int sq0 = blockIdx.y * 128;
  const int sk0 = blockIdx.x * 128;

  const _Float16* qpb = qp + (size_t)b * S_LEN * D_DIM;
  const _Float16* kpb = kp + (size_t)b * S_LEN * D_DIM;

  v8f acc[4][2] = {};
  gemm_body<D_DIM / 32>(qpb + (size_t)sq0 * D_DIM, D_DIM,
                        kpb + (size_t)sk0 * D_DIM, D_DIM,
                        As, Bs, tid, wave_m, wave_n, lane16, h, acc);

  _Float16* lb = logits + (size_t)b * S_LEN * S_LEN;
#pragma unroll
  for (int mt = 0; mt < 4; ++mt)
#pragma unroll
    for (int nt = 0; nt < 2; ++nt) {
      int sk = sk0 + wave_n * 32 + nt * 16 + lane16;
#pragma unroll
      for (int j = 0; j < 8; ++j) {
        int sq = sq0 + wave_m * 64 + mt * 16 + j + 8 * h;
        float val = fmaxf(acc[mt][nt][j], 0.0f) * 0.03125f   // 1/sqrt(1024)
                    + mask[(size_t)sq * S_LEN + sk];
        lb[(size_t)sq * S_LEN + sk] = (_Float16)val;
      }
    }
}

// ---------------------------------------------------------------------------
// Kernel 3: out[b][sq][d] = relu(logits[b] @ vp[b]), K = S_LEN.
// logits f16 [b][S][S]; vpT f16 [b][D][S]; out f32 [b][S][D].
// ---------------------------------------------------------------------------
__global__ __launch_bounds__(256) void out_kernel(const _Float16* __restrict__ logits,
                                                  const _Float16* __restrict__ vpT,
                                                  float* __restrict__ out) {
  __shared__ _Float16 As[2 * TILE_ELEMS];
  __shared__ _Float16 Bs[2 * TILE_ELEMS];

  const int tid = threadIdx.x;
  const int wid = tid >> 5, lane = tid & 31;
  const int lane16 = lane & 15, h = lane >> 4;
  const int wave_m = wid & 1, wave_n = wid >> 1;
  const int b = blockIdx.z;
  const int sq0 = blockIdx.y * 128;
  const int d0 = blockIdx.x * 128;

  const _Float16* lb = logits + (size_t)b * S_LEN * S_LEN;
  const _Float16* vb = vpT + (size_t)b * D_DIM * S_LEN;

  v8f acc[4][2] = {};
  gemm_body<S_LEN / 32>(lb + (size_t)sq0 * S_LEN, S_LEN,
                        vb + (size_t)d0 * S_LEN, S_LEN,
                        As, Bs, tid, wave_m, wave_n, lane16, h, acc);

  float* ob = out + (size_t)b * S_LEN * D_DIM;
#pragma unroll
  for (int mt = 0; mt < 4; ++mt)
#pragma unroll
    for (int nt = 0; nt < 2; ++nt) {
      int d = d0 + wave_n * 32 + nt * 16 + lane16;
#pragma unroll
      for (int j = 0; j < 8; ++j) {
        int sq = sq0 + wave_m * 64 + mt * 16 + j + 8 * h;
        ob[(size_t)sq * D_DIM + d] = fmaxf(acc[mt][nt][j], 0.0f);
      }
    }
}

// ---------------------------------------------------------------------------
// Host launcher
// ---------------------------------------------------------------------------
extern "C" void kernel_launch(void* const* d_in, const int* in_sizes, int n_in,
                              void* d_out, int out_size, void* d_ws, size_t ws_size,
                              hipStream_t stream) {
  (void)in_sizes; (void)n_in; (void)out_size;

  const float* q      = (const float*)d_in[0];
  const float* k      = (const float*)d_in[1];
  const float* v      = (const float*)d_in[2];
  const float* mask   = (const float*)d_in[3];
  const float* Wq     = (const float*)d_in[4];  // [1,1,1024,1024] == [K][N]
  const float* Wk     = (const float*)d_in[5];
  const float* Wv     = (const float*)d_in[6];
  const float* bias_q = (const float*)d_in[7];
  const float* bias_k = (const float*)d_in[8];
  const float* bias_v = (const float*)d_in[9];
  float* out = (float*)d_out;

  // Workspace layout (bytes). logits aliases the xh region (dead after proj).
  const size_t MB = 1024 * 1024;
  const size_t off_Wtq    = 0;         // 2 MB each
  const size_t off_Wtk    = 2 * MB;
  const size_t off_Wtv    = 4 * MB;
  const size_t off_xhq    = 6 * MB;    // 16 MB each, f16 [8192][1024]
  const size_t off_xhk    = 22 * MB;
  const size_t off_xhv    = 38 * MB;
  const size_t off_qp     = 54 * MB;   // 16 MB
  const size_t off_kp     = 70 * MB;   // 16 MB
  const size_t off_vpT    = 86 * MB;   // 16 MB f16 [B][D][S]
  const size_t off_logits = 6 * MB;    // 32 MB f16 [B][S][S], aliases xhq/xhk
  const size_t need       = 102 * MB;
  if (ws_size < need) return;          // workspace too small; bail safely

  char* ws = (char*)d_ws;
  _Float16* Wtq    = (_Float16*)(ws + off_Wtq);
  _Float16* Wtk    = (_Float16*)(ws + off_Wtk);
  _Float16* Wtv    = (_Float16*)(ws + off_Wtv);
  _Float16* xhq    = (_Float16*)(ws + off_xhq);
  _Float16* xhk    = (_Float16*)(ws + off_xhk);
  _Float16* xhv    = (_Float16*)(ws + off_xhv);
  _Float16* qp     = (_Float16*)(ws + off_qp);
  _Float16* kp     = (_Float16*)(ws + off_kp);
  _Float16* vpT    = (_Float16*)(ws + off_vpT);
  _Float16* logits = (_Float16*)(ws + off_logits);

  // 0a) activation f32 -> f16 (streaming)
  {
    dim3 blk(256), grd((M_TOT * D_DIM) / (256 * 8));   // 4096 blocks
    cvt_f16_kernel<<<grd, blk, 0, stream>>>(q, xhq);
    cvt_f16_kernel<<<grd, blk, 0, stream>>>(k, xhk);
    cvt_f16_kernel<<<grd, blk, 0, stream>>>(v, xhv);
  }
  // 0b) weight transpose + f16 convert
  {
    dim3 tb(32, 8), tg(D_DIM / 32, D_DIM / 32);
    transpose_w_kernel<<<tg, tb, 0, stream>>>(Wq, Wtq);
    transpose_w_kernel<<<tg, tb, 0, stream>>>(Wk, Wtk);
    transpose_w_kernel<<<tg, tb, 0, stream>>>(Wv, Wtv);
  }
  // 1) projections
  {
    dim3 blk(256), grd(D_DIM / 128, M_TOT / 128);
    proj_kernel<0><<<grd, blk, 0, stream>>>(xhq, Wtq, bias_q, qp);
    proj_kernel<0><<<grd, blk, 0, stream>>>(xhk, Wtk, bias_k, kp);
    proj_kernel<1><<<grd, blk, 0, stream>>>(xhv, Wtv, bias_v, vpT);
  }
  // 2) attention logits
  {
    dim3 blk(256), grd(S_LEN / 128, S_LEN / 128, BATCH);
    scores_kernel<<<grd, blk, 0, stream>>>(qp, kp, mask, logits);
  }
  // 3) output
  {
    dim3 blk(256), grd(D_DIM / 128, S_LEN / 128, BATCH);
    out_kernel<<<grd, blk, 0, stream>>>(logits, vpT, out);
  }
}
